// RPE_59133109731847
// MI455X (gfx1250) — matrix operand order repaired and checked by reading, not compiled, not run on hardware
//
#include <hip/hip_runtime.h>
#include <stdint.h>

#define LATENT  512
#define NLAYERS 8
#define ODIM    64
#define MT      64      // rows per workgroup
#define TPB     256     // 8 wave32s

typedef __bf16 v8bf  __attribute__((ext_vector_type(8)));
typedef __bf16 v16bf __attribute__((ext_vector_type(16)));
typedef float  v4f   __attribute__((ext_vector_type(4)));
typedef float  v8f   __attribute__((ext_vector_type(8)));

union BF16Frag { v16bf v; v8bf h[2]; };

// ---- cheap tanh: hardware V_TANH_F32 when available, else exp2+rcp ----
__device__ __forceinline__ float fast_tanh(float u) {
#if __has_builtin(__builtin_amdgcn_tanhf)
    return __builtin_amdgcn_tanhf(u);
#else
    // tanh(u) = 1 - 2/(exp(2u)+1); exp(2u) = exp2(u * 2*log2(e))
    float e = __builtin_amdgcn_exp2f(u * 2.8853900817779268f);
    return 1.0f - 2.0f * __builtin_amdgcn_rcpf(e + 1.0f);
#endif
}

__device__ __forceinline__ float gelu_tanh(float x) {
    const float k0 = 0.7978845608028654f;   // sqrt(2/pi)
    const float k1 = 0.044715f;
    float inner = k0 * (x + k1 * x * x * x);
    return 0.5f * x * (1.0f + fast_tanh(inner));
}

// ---- one-time f32 -> bf16 weight conversion into workspace ----
__global__ void wconvert_kernel(const float* __restrict__ lin_w,
                                const float* __restrict__ out_w,
                                __bf16* __restrict__ ws) {
    int i = blockIdx.x * blockDim.x + threadIdx.x;
    const int NL = NLAYERS * LATENT * LATENT;
    if (i < NL) {
        ws[i] = (__bf16)lin_w[i];
    } else {
        int j = i - NL;
        if (j < ODIM * LATENT) ws[NL + j] = (__bf16)out_w[j];
    }
}

// ---- CDNA5 async global->LDS staging (ASYNCcnt-tracked) ----
// Stages one [512 x 32] bf16 K-panel: 2048 16B chunks, 8 async b128 per thread.
__device__ __forceinline__ void stage_panel_async(const __bf16* wl, int p,
                                                  __bf16 (*wbuf)[40], int tid) {
    const int n0   = tid >> 2;            // base row
    const int part = (tid & 3) * 8;       // bf16 element offset within panel row
    const uint64_t base = (uint64_t)(uintptr_t)wl;
    #pragma unroll
    for (int k = 0; k < 8; ++k) {
        int n = n0 + k * 64;
        uint32_t voff = (uint32_t)(n * LATENT + p * 32 + part) * 2u;  // bytes
        // LDS aperture: flat addr[31:0] == LDS byte address (ISA sec.10.2)
        uint32_t ldsa = (uint32_t)(uintptr_t)&wbuf[n][part];
        asm volatile("global_load_async_to_lds_b128 %0, %1, %2"
                     :: "v"(ldsa), "v"(voff), "s"(base)
                     : "memory");
    }
}

__device__ __forceinline__ void wait_async0() {
    asm volatile("s_wait_asynccnt 0x0" ::: "memory");
}

// ---- LayerNorm + GELU: h_s (f32) -> z_s (bf16); 4 lanes per row ----
__device__ __forceinline__ void ln_gelu(const float (*h)[LATENT + 4],
                                        __bf16 (*z)[LATENT + 8],
                                        const float* __restrict__ w,
                                        const float* __restrict__ b,
                                        int tid) {
    const int row = tid >> 2;
    const int c0  = (tid & 3) * 128;
    const v4f* hv = (const v4f*)&h[row][c0];   // 16B aligned (row stride 516 f32)

    float s = 0.0f, ss = 0.0f;
    #pragma unroll 8
    for (int j = 0; j < 32; ++j) {
        v4f v = hv[j];
        s  += v[0] + v[1] + v[2] + v[3];
        ss += v[0] * v[0] + v[1] * v[1] + v[2] * v[2] + v[3] * v[3];
    }
    s  += __shfl_xor(s, 1, 32);  s  += __shfl_xor(s, 2, 32);
    ss += __shfl_xor(ss, 1, 32); ss += __shfl_xor(ss, 2, 32);
    const float mu   = s * (1.0f / LATENT);
    const float var  = ss * (1.0f / LATENT) - mu * mu;
    const float rstd = rsqrtf(var + 1e-5f);

    #pragma unroll 2
    for (int jb = 0; jb < 16; ++jb) {          // 8 columns per iteration
        v4f a0 = hv[2 * jb], a1 = hv[2 * jb + 1];
        v8bf o8;
        #pragma unroll
        for (int e = 0; e < 4; ++e) {
            int c = c0 + jb * 8 + e;
            o8[e]     = (__bf16)gelu_tanh((a0[e] - mu) * rstd * w[c]     + b[c]);
            o8[e + 4] = (__bf16)gelu_tanh((a1[e] - mu) * rstd * w[c + 4] + b[c + 4]);
        }
        *(v8bf*)&z[row][c0 + jb * 8] = o8;
    }
}

// ---- fused MLP: projection -> 8 residual blocks -> output projection ----
__global__ __launch_bounds__(TPB) void mlp_kernel(
    const float* __restrict__ x,
    const float* __restrict__ proj_w, const float* __restrict__ proj_b,
    const float* __restrict__ ln_w,   const float* __restrict__ ln_b,
    const float* __restrict__ out_ln_w, const float* __restrict__ out_ln_b,
    const __bf16* __restrict__ wb,      // [8][512][512] bf16
    const __bf16* __restrict__ out_wb,  // [64][512] bf16
    float* __restrict__ out) {

    __shared__ float  h_s[MT][LATENT + 4];     // 129 KB, bank-spread
    __shared__ __bf16 z_s[MT][LATENT + 8];     //  65 KB
    __shared__ __bf16 w_s[2][LATENT][40];      //  80 KB: double-buffered K panels

    const int tid = threadIdx.x;
    const int r0  = blockIdx.x * MT;

    // --- projection: h[m][c] = x[r0+m] * proj_w[c] + proj_b[c] ---
    for (int idx = tid; idx < MT * LATENT; idx += TPB) {
        int m = idx >> 9, c = idx & (LATENT - 1);
        h_s[m][c] = x[r0 + m] * proj_w[c] + proj_b[c];
    }

    // wave tiling: wave -> 2 m-tiles x 8 n-tiles (B fragments feed 2 WMMAs)
    const int wv   = tid >> 5;
    const int lane = tid & 31;
    const int mg   = wv >> 2;          // m-group: rows [32*mg, 32*mg+32)
    const int nq   = wv & 3;           // n-quarter: cols [128*nq, 128*nq+128)
    const int nl   = lane & 15;        // n_local (B/C/D) == m_local (A)
    const int hi   = lane >> 4;        // lane half
    const int mrow0 = mg * 32 + hi * 8;        // C/D rows, m-tile 0
    const int mrow1 = mrow0 + 16;              // C/D rows, m-tile 1
    const int am0   = mg * 32 + nl;            // A source row, m-tile 0
    const int am1   = am0 + 16;                // A source row, m-tile 1

    for (int layer = 0; layer < NLAYERS; ++layer) {
        __syncthreads();   // h stable, prev layer's z/w consumers done
        ln_gelu(h_s, z_s, ln_w + layer * LATENT, ln_b + layer * LATENT, tid);

        // load C = current h tile (residual accumulate for free)
        v8f acc[2][8];
        #pragma unroll
        for (int nt = 0; nt < 8; ++nt) {
            int col = nq * 128 + nt * 16 + nl;
            #pragma unroll
            for (int v = 0; v < 8; ++v) {
                acc[0][nt][v] = h_s[mrow0 + v][col];
                acc[1][nt][v] = h_s[mrow1 + v][col];
            }
        }

        const __bf16* wl = wb + (size_t)layer * LATENT * LATENT;
        stage_panel_async(wl, 0, w_s[0], tid);     // prologue panel
        wait_async0();
        __syncthreads();                           // panel 0 + z_s ready

        #pragma unroll 1
        for (int p = 0; p < 16; ++p) {             // 16 K-panels of width 32
            const int buf = p & 1;
            if (p < 15) stage_panel_async(wl, p + 1, w_s[buf ^ 1], tid);

            BF16Frag A0, A1;  // A 16x32: K chunks {0..7,16..23}/{8..15,24..31}
            A0.h[0] = *(const v8bf*)&z_s[am0][p * 32 + hi * 8];
            A0.h[1] = *(const v8bf*)&z_s[am0][p * 32 + hi * 8 + 16];
            A1.h[0] = *(const v8bf*)&z_s[am1][p * 32 + hi * 8];
            A1.h[1] = *(const v8bf*)&z_s[am1][p * 32 + hi * 8 + 16];

            #pragma unroll
            for (int nt = 0; nt < 8; ++nt) {
                int n = nq * 128 + nt * 16 + nl;   // W row = B column
                BF16Frag B;  // B 32x16: K chunks {0..7,16..23}/{8..15,24..31}
                B.h[0] = *(const v8bf*)&w_s[buf][n][hi * 8];
                B.h[1] = *(const v8bf*)&w_s[buf][n][hi * 8 + 16];
                acc[0][nt] = __builtin_amdgcn_wmma_f32_16x16x32_bf16(
                    false, A0.v, false, B.v, (short)0, acc[0][nt], false, false);
                acc[1][nt] = __builtin_amdgcn_wmma_f32_16x16x32_bf16(
                    false, A1.v, false, B.v, (short)0, acc[1][nt], false, false);
            }
            wait_async0();      // next panel landed in LDS
            __syncthreads();    // all waves done with buf / writes to buf^1 visible
        }

        // write back h += z @ W^T (each element owned by exactly one lane)
        #pragma unroll
        for (int nt = 0; nt < 8; ++nt) {
            int col = nq * 128 + nt * 16 + nl;
            #pragma unroll
            for (int v = 0; v < 8; ++v) {
                h_s[mrow0 + v][col] = acc[0][nt][v];
                h_s[mrow1 + v][col] = acc[1][nt][v];
            }
        }
    }

    // --- output: LN -> GELU -> [512 -> 64] projection ---
    __syncthreads();
    ln_gelu(h_s, z_s, out_ln_w, out_ln_b, tid);
    __syncthreads();

    const int mi = wv >> 1;            // 4 m-tiles
    const int np = wv & 1;             // 2 output n-tiles each
    const int omrow = mi * 16 + hi * 8;
    const int oam   = mi * 16 + nl;
    v8f oacc[2];
    #pragma unroll
    for (int j = 0; j < 2; ++j)
        #pragma unroll
        for (int v = 0; v < 8; ++v) oacc[j][v] = 0.0f;

    #pragma unroll 1
    for (int kk = 0; kk < 16; ++kk) {
        BF16Frag A;
        A.h[0] = *(const v8bf*)&z_s[oam][kk * 32 + hi * 8];
        A.h[1] = *(const v8bf*)&z_s[oam][kk * 32 + hi * 8 + 16];
        #pragma unroll
        for (int j = 0; j < 2; ++j) {
            int n = (np * 2 + j) * 16 + nl;
            BF16Frag B;   // out_w is tiny & L2/WGP$-resident: read from global
            B.h[0] = *(const v8bf*)(out_wb + n * LATENT + kk * 32 + hi * 8);
            B.h[1] = *(const v8bf*)(out_wb + n * LATENT + kk * 32 + hi * 8 + 16);
            oacc[j] = __builtin_amdgcn_wmma_f32_16x16x32_bf16(
                false, A.v, false, B.v, (short)0, oacc[j], false, false);
        }
    }

    #pragma unroll
    for (int j = 0; j < 2; ++j) {
        int n0 = (np * 2 + j) * 16 + nl;
        #pragma unroll
        for (int v = 0; v < 8; ++v)
            out[(size_t)(r0 + omrow + v) * ODIM + n0] = oacc[j][v];
    }
}

extern "C" void kernel_launch(void* const* d_in, const int* in_sizes, int n_in,
                              void* d_out, int out_size, void* d_ws, size_t ws_size,
                              hipStream_t stream) {
    (void)n_in; (void)out_size; (void)ws_size;
    const float* x        = (const float*)d_in[0];
    const float* proj_w   = (const float*)d_in[1];
    const float* proj_b   = (const float*)d_in[2];
    const float* ln_w     = (const float*)d_in[3];
    const float* ln_b     = (const float*)d_in[4];
    const float* lin_w    = (const float*)d_in[5];
    const float* out_ln_w = (const float*)d_in[6];
    const float* out_ln_b = (const float*)d_in[7];
    const float* out_w    = (const float*)d_in[8];
    float*  out = (float*)d_out;
    __bf16* ws  = (__bf16*)d_ws;

    const int NL    = NLAYERS * LATENT * LATENT;
    const int total = NL + ODIM * LATENT;
    wconvert_kernel<<<(total + TPB - 1) / TPB, TPB, 0, stream>>>(lin_w, out_w, ws);

    const int N = in_sizes[0];   // x is [N,1]
    mlp_kernel<<<N / MT, TPB, 0, stream>>>(x, proj_w, proj_b, ln_w, ln_b,
                                           out_ln_w, out_ln_b, ws, ws + NL, out);
}
